// MTALSTM_21706764714056
// MI455X (gfx1250) — matrix operand order repaired and checked by reading, not compiled, not run on hardware
//
#include <hip/hip_runtime.h>
#include <stdint.h>

// ---------------- problem constants ----------------
#define Bv 32
#define Sv 64
#define Ev 512
#define Hv 512
#define Kv 5
#define C1v 2500
#define C2v 10000
#define CSZ1v (C2v - C1v)      // 7500
#define CSZ2v (50000 - C2v)    // 40000
#define H1v 128
#define H2v 32

#define Nrows (Sv * Bv)        // 2048
#define OFF_LOSS  (Nrows)
#define OFF_OUTS  (Nrows + 1)
#define OFF_HF    (OFF_OUTS + Sv * Bv * Hv)
#define OFF_CF    (OFF_HF + 2 * Bv * Hv)
#define OFF_ATTN  (OFF_CF + 2 * Bv * Hv)
#define OFF_COV   (OFF_ATTN + Sv * Bv * Kv)

typedef __attribute__((ext_vector_type(16))) __bf16 v16bf;
typedef __attribute__((ext_vector_type(8)))  __bf16 v8bf;
typedef __attribute__((ext_vector_type(8)))  float  v8f;

// ---------------- WMMA helpers (CDNA5 bf16 16x16x32, fp32 accum) -------------
static __device__ inline v8f wmma_bf16(v16bf a, v16bf b, v8f c) {
  return __builtin_amdgcn_wmma_f32_16x16x32_bf16(false, a, false, b, (short)0, c,
                                                 false, false);
}

// A fragment (16x32 bf16, row-major source): lane holds row M=lane&15,
// K = {k0..k0+7, k0+16..k0+23}, k0 = (lane>=16)?8:0  (ISA 7.12.2)
static __device__ inline v16bf load_afrag(const __bf16* p, int half) {
  v8bf lo = *(const v8bf*)(p + half * 8);
  v8bf hi = *(const v8bf*)(p + 16 + half * 8);
  return __builtin_shufflevector(lo, hi, 0, 1, 2, 3, 4, 5, 6, 7, 8, 9, 10, 11,
                                 12, 13, 14, 15);
}
// B fragment (32x16): lane holds column N=lane&15 = weight row;
// K = {k0..k0+7, k0+16..k0+23}, k0 = (lane>=16)?8:0 — weights stored row-major [N, K] (we do x @ W^T).
static __device__ inline v16bf load_bfrag(const __bf16* p, int half) {
  v8bf lo = *(const v8bf*)(p + half * 8);
  v8bf hi = *(const v8bf*)(p + 16 + half * 8);
  return __builtin_shufflevector(lo, hi, 0, 1, 2, 3, 4, 5, 6, 7, 8, 9, 10, 11,
                                 12, 13, 14, 15);
}

// One 16x16 output tile of D = A[Mx KD] @ W[N x KD]^T
template <int KD>
static __device__ inline v8f tile_mm(const __bf16* A, const __bf16* W, int mt,
                                     int nt, v8f acc, int lane) {
  const int half = (lane >> 4) & 1;
  const int rc = lane & 15;
  const __bf16* arow = A + (size_t)(mt * 16 + rc) * KD;
  const __bf16* wrow = W + (size_t)(nt * 16 + rc) * KD;
#pragma unroll
  for (int kb = 0; kb < KD / 32; ++kb) {
    v16bf a = load_afrag(arow + kb * 32, half);
    v16bf b = load_bfrag(wrow + kb * 32, half);
    acc = wmma_bf16(a, b, acc);
  }
  return acc;
}

// C/D fragment layout: VGPR v -> row mt*16 + v + 8*(lane>=16), col = nt*16+(lane&15)
static __device__ inline void store_tile(float* D, int ldd, int mt, int nt,
                                         v8f acc, int lane) {
  const int half = (lane >> 4) & 1;
  const int rc = lane & 15;
  const int r0 = mt * 16 + half * 8;
  const int c = nt * 16 + rc;
#pragma unroll
  for (int v = 0; v < 8; ++v) D[(size_t)(r0 + v) * ldd + c] = acc[v];
}

static __device__ inline float sigf(float x) { return 1.f / (1.f + expf(-x)); }

// ---------------- generic GEMM: D = A1@W1^T (+ A2@W2^T) ----------------------
template <int KD1, int KD2>
__global__ void gemm2_kernel(const __bf16* A1, const __bf16* W1,
                             const __bf16* A2, const __bf16* W2, float* D,
                             int Mt, int Nt, int ldd) {
  const int wid = (blockIdx.x * blockDim.x + threadIdx.x) >> 5;
  const int lane = threadIdx.x & 31;
  if (wid >= Mt * Nt) return;  // wave-uniform: EXEC stays full for WMMA
  const int mt = wid / Nt, nt = wid % Nt;
  v8f acc = {};
  acc = tile_mm<KD1>(A1, W1, mt, nt, acc, lane);
  if (KD2 > 0) acc = tile_mm<KD2>(A2, W2, mt, nt, acc, lane);
  store_tile(D, ldd, mt, nt, acc, lane);
}

// ---------------- weight fp32 -> bf16 (with optional zero row padding) -------
__global__ void cvt_pad_kernel(__bf16* dst, const float* src, int n_src,
                               int n_total) {
  int idx = blockIdx.x * blockDim.x + threadIdx.x;
  if (idx >= n_total) return;
  dst[idx] = (idx < n_src) ? (__bf16)src[idx] : (__bf16)0.f;
}

// ---------------- gathers ----------------------------------------------------
__global__ void gather_x_kernel(const int* inputs, const float* emb,
                                __bf16* x_bf) {
  const int g = blockIdx.x;  // g = t*B + b
  const int t = g >> 5, b = g & 31;
  const int row = inputs[b * Sv + t];
  const float* src = emb + (size_t)row * Ev;
  __bf16* dst = x_bf + (size_t)g * (2 * Ev);
  for (int e = threadIdx.x; e < Ev; e += blockDim.x) dst[e] = (__bf16)src[e];
}

__global__ void gather_topics_kernel(const int* topics, const float* emb,
                                     float* tf, __bf16* tb) {
  const int g = blockIdx.x;  // g = b*K + k
  const int row = topics[g];
  const float* src = emb + (size_t)row * Ev;
  for (int e = threadIdx.x; e < Ev; e += blockDim.x) {
    float v = src[e];
    tf[(size_t)g * Ev + e] = v;
    tb[(size_t)g * Ev + e] = (__bf16)v;
  }
}

// ---------------- state init -------------------------------------------------
__global__ void init_kernel(const float* outp, const float* h0in,
                            const float* c0in, const float* covin, float* q,
                            float* h0f, float* h1f, float* c0f, float* c1f,
                            __bf16* h0bf, __bf16* h1bf, float* cov) {
  int idx = blockIdx.x * blockDim.x + threadIdx.x;
  if (idx < Bv * Hv) {
    q[idx] = outp[idx];
    float a = h0in[idx], b2 = h0in[Bv * Hv + idx];
    h0f[idx] = a; h1f[idx] = b2;
    h0bf[idx] = (__bf16)a; h1bf[idx] = (__bf16)b2;
    c0f[idx] = c0in[idx];
    c1f[idx] = c0in[Bv * Hv + idx];
  }
  if (idx < Bv * Kv) cov[idx] = covin[idx];
}

// ---------------- phi[b,k] = sum(mask[b]) * sigmoid(topics_flat[b] . Uf[k]) --
__global__ void phi_kernel(const float* topics_f, const float* Uf,
                           const float* mask, float* phi) {
  const int g = blockIdx.x;  // b*K + k
  const int b = g / Kv, k = g % Kv;
  const int lane = threadIdx.x;
  const float* tb = topics_f + (size_t)b * Kv * Ev;
  const float* ur = Uf + (size_t)k * (Kv * Ev);
  float part = 0.f;
  for (int e = lane; e < Kv * Ev; e += 32) part += tb[e] * ur[e];
  for (int off = 16; off >= 1; off >>= 1) part += __shfl_down(part, off, 32);
  if (lane == 0) {
    float ms = 0.f;
    for (int s2 = 0; s2 < Sv; ++s2) ms += mask[b * Sv + s2];
    phi[g] = ms * sigf(part);
  }
}

// ---------------- one attention step: qq-chain, softmax, mt, cov -------------
__global__ __launch_bounds__(1024) void attention_kernel(
    int t, const float* q_buf, float* qqA, __bf16* qq_bf, const __bf16* Wa_bf,
    const float* projk, const float* va_w, const float* va_b, float* cov,
    const float* phi, const float* topics_f, __bf16* x_bf, float* attn_out) {
  __shared__ float sc[Bv * Kv];
  __shared__ float al[Bv * Kv];
  const int tid = threadIdx.x;
  const int wid = tid >> 5, lane = tid & 31;

  for (int e = tid; e < Bv * Hv; e += 1024) qq_bf[e] = (__bf16)q_buf[e];
  __threadfence();
  __syncthreads();

  for (int i = 0; i < Kv; ++i) {
    // qqA = qq_bf @ Wa^T : 2 x 32 tiles over 32 waves
    for (int tile = wid; tile < 64; tile += 32) {
      const int mt = tile >> 5, nt = tile & 31;
      v8f acc = {};
      acc = tile_mm<Hv>(qq_bf, Wa_bf, mt, nt, acc, lane);
      store_tile(qqA, Hv, mt, nt, acc, lane);
    }
    __threadfence();
    __syncthreads();
    // score for row b = wid: (tanh(qq+pk[:,i]) . va + va_b) * cov[:,i]
    {
      const int b = wid;
      const float* row = qqA + (size_t)b * Hv;
      const float* pk = projk + ((size_t)b * Kv + i) * Hv;
      float part = 0.f;
      for (int e = lane; e < Hv; e += 32)
        part += va_w[e] * tanhf(row[e] + pk[e]);
      for (int off = 16; off >= 1; off >>= 1) part += __shfl_down(part, off, 32);
      if (lane == 0) sc[b * Kv + i] = (part + va_b[0]) * cov[b * Kv + i];
    }
    // bf16 copy for next chain iteration
    for (int e = tid; e < Bv * Hv; e += 1024) qq_bf[e] = (__bf16)qqA[e];
    __threadfence();
    __syncthreads();
  }

  if (tid < Bv) {
    const int b = tid;
    float m = sc[b * Kv];
    for (int k = 1; k < Kv; ++k) m = fmaxf(m, sc[b * Kv + k]);
    float ssum = 0.f, ex[Kv];
    for (int k = 0; k < Kv; ++k) { ex[k] = expf(sc[b * Kv + k] - m); ssum += ex[k]; }
    for (int k = 0; k < Kv; ++k) {
      float a = ex[k] / ssum;
      al[b * Kv + k] = a;
      attn_out[b * Kv + k] = a;
      cov[b * Kv + k] = cov[b * Kv + k] - a / phi[b * Kv + k];
    }
  }
  __syncthreads();

  // mt[b,e] = sum_k alpha[b,k]*topics_e[b,k,e] -> x_bf[t][b][512+e]
  __bf16* xdst = x_bf + ((size_t)t * Bv) * (2 * Ev) + Ev;
  for (int e = tid; e < Bv * Ev; e += 1024) {
    const int b = e >> 9, j = e & 511;
    float mv = 0.f;
    for (int k = 0; k < Kv; ++k)
      mv += al[b * Kv + k] * topics_f[((size_t)b * Kv + k) * Ev + j];
    xdst[(size_t)b * (2 * Ev) + j] = (__bf16)mv;
  }
}

// ---------------- LSTM elementwise (gate order i,f,g,o) ----------------------
__global__ void lstm_elem_kernel(const float* gates, const float* bih,
                                 const float* bhh, float* cst, float* hst,
                                 __bf16* hbf, float* outs_dst,
                                 __bf16* outsbf_dst, float* q_dst) {
  const int idx = blockIdx.x * blockDim.x + threadIdx.x;
  if (idx >= Bv * Hv) return;
  const int b = idx >> 9, j = idx & 511;
  const float* g = gates + (size_t)b * 4 * Hv;
  float gi = g[j] + bih[j] + bhh[j];
  float gf = g[Hv + j] + bih[Hv + j] + bhh[Hv + j];
  float gg = g[2 * Hv + j] + bih[2 * Hv + j] + bhh[2 * Hv + j];
  float go = g[3 * Hv + j] + bih[3 * Hv + j] + bhh[3 * Hv + j];
  float c = sigf(gf) * cst[idx] + sigf(gi) * tanhf(gg);
  float h = sigf(go) * tanhf(c);
  cst[idx] = c; hst[idx] = h; hbf[idx] = (__bf16)h;
  if (outs_dst) {
    outs_dst[idx] = h;
    outsbf_dst[idx] = (__bf16)h;
    q_dst[idx] = h;
  }
}

// ---------------- streaming log-sum-exp over C classes via WMMA --------------
template <int KD>
__global__ void lse_kernel(const __bf16* X, const __bf16* W, int Ct, int C,
                           float* lse) {
  const int lane = threadIdx.x & 31;
  const int rowt = blockIdx.x;
  const int half = (lane >> 4) & 1, rc = lane & 15;
  const __bf16* arow = X + (size_t)(rowt * 16 + rc) * KD;
  v16bf afrag[KD / 32];
#pragma unroll
  for (int kb = 0; kb < KD / 32; ++kb)
    afrag[kb] = load_afrag(arow + kb * 32, half);
  float m[8], s[8];
#pragma unroll
  for (int v = 0; v < 8; ++v) { m[v] = -1e30f; s[v] = 0.f; }
  for (int ct = 0; ct < Ct; ++ct) {
    v8f acc = {};
    const __bf16* wrow = W + (size_t)(ct * 16 + rc) * KD;
#pragma unroll
    for (int kb = 0; kb < KD / 32; ++kb)
      acc = wmma_bf16(afrag[kb], load_bfrag(wrow + kb * 32, half), acc);
    const int col = ct * 16 + rc;
    if (col < C) {  // mask pad columns
#pragma unroll
      for (int v = 0; v < 8; ++v) {
        float val = acc[v];
        if (val > m[v]) { s[v] = s[v] * expf(m[v] - val) + 1.f; m[v] = val; }
        else s[v] += expf(val - m[v]);
      }
    }
  }
#pragma unroll
  for (int v = 0; v < 8; ++v) {  // reduce over 16 lanes holding same row
    float mm = m[v], ss = s[v];
    for (int off = 8; off >= 1; off >>= 1) {
      float om = __shfl_xor(mm, off, 32);
      float os = __shfl_xor(ss, off, 32);
      float nm = fmaxf(mm, om);
      ss = ss * expf(mm - nm) + os * expf(om - nm);
      mm = nm;
    }
    if (rc == 0) lse[rowt * 16 + half * 8 + v] = mm + logf(ss);
  }
}

// ---------------- target logits (dot products) -------------------------------
__global__ void tgt_kernel(const int* target, const __bf16* xbf,
                           const __bf16* headbf, const __bf16* h1bf,
                           const __bf16* t1w2bf, const __bf16* h2bf,
                           const __bf16* t2w2bf, float* hdt, float* hdc1,
                           float* hdc2, float* l1t, float* l2t) {
  const int wid = (blockIdx.x * blockDim.x + threadIdx.x) >> 5;
  const int lane = threadIdx.x & 31;
  if (wid >= Nrows) return;
  const int n = wid, b = n & (Bv - 1), s = n >> 5;
  const int tg = target[b * Sv + s];
  const __bf16* x = xbf + (size_t)n * Hv;
  int tr = tg < 0 ? 0 : (tg > C1v - 1 ? C1v - 1 : tg);
  const __bf16* r0 = headbf + (size_t)tr * Hv;
  const __bf16* r1 = headbf + (size_t)C1v * Hv;
  const __bf16* r2 = headbf + (size_t)(C1v + 1) * Hv;
  float d0 = 0.f, d1 = 0.f, d2 = 0.f;
  for (int e = lane; e < Hv; e += 32) {
    float xv = (float)x[e];
    d0 += xv * (float)r0[e];
    d1 += xv * (float)r1[e];
    d2 += xv * (float)r2[e];
  }
  int t1i = tg - C1v; t1i = t1i < 0 ? 0 : (t1i > CSZ1v - 1 ? CSZ1v - 1 : t1i);
  const __bf16* h1 = h1bf + (size_t)n * H1v;
  const __bf16* w1r = t1w2bf + (size_t)t1i * H1v;
  float d3 = 0.f;
  for (int e = lane; e < H1v; e += 32) d3 += (float)h1[e] * (float)w1r[e];
  int t2i = tg - C2v; t2i = t2i < 0 ? 0 : (t2i > CSZ2v - 1 ? CSZ2v - 1 : t2i);
  const __bf16* h2 = h2bf + (size_t)n * H2v;
  const __bf16* w2r = t2w2bf + (size_t)t2i * H2v;
  float d4 = (float)h2[lane] * (float)w2r[lane];  // H2v == 32
  for (int off = 16; off >= 1; off >>= 1) {
    d0 += __shfl_down(d0, off, 32);
    d1 += __shfl_down(d1, off, 32);
    d2 += __shfl_down(d2, off, 32);
    d3 += __shfl_down(d3, off, 32);
    d4 += __shfl_down(d4, off, 32);
  }
  if (lane == 0) { hdt[n] = d0; hdc1[n] = d1; hdc2[n] = d2; l1t[n] = d3; l2t[n] = d4; }
}

// ---------------- combine, out_lp, loss --------------------------------------
__global__ __launch_bounds__(1024) void final_kernel(
    const int* target, const float* hdt, const float* hdc1, const float* hdc2,
    const float* l1t, const float* l2t, const float* lseh, const float* lse1,
    const float* lse2, float* out) {
  __shared__ float red[1024];
  const int tid = threadIdx.x;
  float local = 0.f;
  for (int n = tid; n < Nrows; n += 1024) {
    const int b = n & 31, s = n >> 5;
    const int tg = target[b * Sv + s];
    float oh = hdt[n] - lseh[n];
    float o1 = (hdc1[n] - lseh[n]) + (l1t[n] - lse1[n]);
    float o2 = (hdc2[n] - lseh[n]) + (l2t[n] - lse2[n]);
    float lp = (tg < C1v) ? oh : ((tg < C2v) ? o1 : o2);
    out[n] = lp;
    local += lp;
  }
  red[tid] = local;
  for (int off = 512; off >= 1; off >>= 1) {
    __syncthreads();
    if (tid < off) red[tid] += red[tid + off];
  }
  __syncthreads();
  if (tid == 0) out[OFF_LOSS] = -red[0] / (float)Nrows;
}

__global__ void copyfinal_kernel(const float* h0f, const float* h1f,
                                 const float* c0f, const float* c1f,
                                 const float* cov, float* dout) {
  int idx = blockIdx.x * blockDim.x + threadIdx.x;
  if (idx < Bv * Hv) {
    dout[OFF_HF + idx] = h0f[idx];
    dout[OFF_HF + Bv * Hv + idx] = h1f[idx];
    dout[OFF_CF + idx] = c0f[idx];
    dout[OFF_CF + Bv * Hv + idx] = c1f[idx];
  }
  if (idx < Bv * Kv) dout[OFF_COV + idx] = cov[idx];
}

// ======================= host launcher =======================================
extern "C" void kernel_launch(void* const* d_in, const int* in_sizes, int n_in,
                              void* d_out, int out_size, void* d_ws,
                              size_t ws_size, hipStream_t stream) {
  (void)in_sizes; (void)n_in; (void)out_size; (void)ws_size;
  const int* inputs = (const int*)d_in[0];
  const int* topics = (const int*)d_in[1];
  const float* output0 = (const float*)d_in[2];
  const float* h0in = (const float*)d_in[3];
  const float* c0in = (const float*)d_in[4];
  const float* mask = (const float*)d_in[5];
  const int* target = (const int*)d_in[6];
  const float* covin = (const float*)d_in[7];
  const float* emb = (const float*)d_in[8];
  const float* Uf = (const float*)d_in[9];
  const float* Ua = (const float*)d_in[10];
  const float* Wa = (const float*)d_in[11];
  const float* va = (const float*)d_in[12];
  const float* vab = (const float*)d_in[13];
  const float* wih0 = (const float*)d_in[14];
  const float* whh0 = (const float*)d_in[15];
  const float* bih0 = (const float*)d_in[16];
  const float* bhh0 = (const float*)d_in[17];
  const float* wih1 = (const float*)d_in[18];
  const float* whh1 = (const float*)d_in[19];
  const float* bih1 = (const float*)d_in[20];
  const float* bhh1 = (const float*)d_in[21];
  const float* headw = (const float*)d_in[22];
  const float* t1w1 = (const float*)d_in[23];
  const float* t1w2 = (const float*)d_in[24];
  const float* t2w1 = (const float*)d_in[25];
  const float* t2w2 = (const float*)d_in[26];
  float* out = (float*)d_out;

  // bump allocator over d_ws, 256B aligned (keeps 16B vector loads aligned)
  uintptr_t base = (uintptr_t)d_ws;
  auto alloc = [&](size_t bytes) -> void* {
    uintptr_t p = base;
    base += (bytes + 255) & ~(size_t)255;
    return (void*)p;
  };
  __bf16* Wa_bf = (__bf16*)alloc((size_t)512 * 512 * 2);
  __bf16* Ua_bf = (__bf16*)alloc((size_t)512 * 512 * 2);
  __bf16* wih0_bf = (__bf16*)alloc((size_t)2048 * 1024 * 2);
  __bf16* whh0_bf = (__bf16*)alloc((size_t)2048 * 512 * 2);
  __bf16* wih1_bf = (__bf16*)alloc((size_t)2048 * 512 * 2);
  __bf16* whh1_bf = (__bf16*)alloc((size_t)2048 * 512 * 2);
  __bf16* head_bf = (__bf16*)alloc((size_t)2512 * 512 * 2);   // pad 2502->2512
  __bf16* t1w1_bf = (__bf16*)alloc((size_t)128 * 512 * 2);
  __bf16* t1w2_bf = (__bf16*)alloc((size_t)7504 * 128 * 2);   // pad 7500->7504
  __bf16* t2w1_bf = (__bf16*)alloc((size_t)32 * 512 * 2);
  __bf16* t2w2_bf = (__bf16*)alloc((size_t)40000 * 32 * 2);
  __bf16* x_bf = (__bf16*)alloc((size_t)Sv * Bv * 1024 * 2);
  float* topics_f = (float*)alloc((size_t)160 * 512 * 4);
  __bf16* topics_bf = (__bf16*)alloc((size_t)160 * 512 * 2);
  float* projk = (float*)alloc((size_t)160 * 512 * 4);
  float* phi = (float*)alloc(160 * 4);
  float* q_buf = (float*)alloc((size_t)Bv * Hv * 4);
  float* qqA = (float*)alloc((size_t)Bv * Hv * 4);
  __bf16* qq_bf = (__bf16*)alloc((size_t)Bv * Hv * 2);
  float* h0f = (float*)alloc((size_t)Bv * Hv * 4);
  float* h1f = (float*)alloc((size_t)Bv * Hv * 4);
  float* c0f = (float*)alloc((size_t)Bv * Hv * 4);
  float* c1f = (float*)alloc((size_t)Bv * Hv * 4);
  __bf16* h0bf = (__bf16*)alloc((size_t)Bv * Hv * 2);
  __bf16* h1bf = (__bf16*)alloc((size_t)Bv * Hv * 2);
  float* cov = (float*)alloc(160 * 4);
  float* gates0 = (float*)alloc((size_t)Bv * 2048 * 4);
  float* gates1 = (float*)alloc((size_t)Bv * 2048 * 4);
  __bf16* outs_bf = (__bf16*)alloc((size_t)Nrows * Hv * 2);
  float* hid1 = (float*)alloc((size_t)Nrows * H1v * 4);
  __bf16* hid1_bf = (__bf16*)alloc((size_t)Nrows * H1v * 2);
  float* hid2 = (float*)alloc((size_t)Nrows * H2v * 4);
  __bf16* hid2_bf = (__bf16*)alloc((size_t)Nrows * H2v * 2);
  float* hdt = (float*)alloc(Nrows * 4);
  float* hdc1 = (float*)alloc(Nrows * 4);
  float* hdc2 = (float*)alloc(Nrows * 4);
  float* l1t = (float*)alloc(Nrows * 4);
  float* l2t = (float*)alloc(Nrows * 4);
  float* lseh = (float*)alloc(Nrows * 4);
  float* lse1 = (float*)alloc(Nrows * 4);
  float* lse2 = (float*)alloc(Nrows * 4);

  auto cvt = [&](__bf16* dst, const float* src, int n_src, int n_total) {
    cvt_pad_kernel<<<(n_total + 255) / 256, 256, 0, stream>>>(dst, src, n_src,
                                                              n_total);
  };
  // weights -> bf16
  cvt(Wa_bf, Wa, 512 * 512, 512 * 512);
  cvt(Ua_bf, Ua, 512 * 512, 512 * 512);
  cvt(wih0_bf, wih0, 2048 * 1024, 2048 * 1024);
  cvt(whh0_bf, whh0, 2048 * 512, 2048 * 512);
  cvt(wih1_bf, wih1, 2048 * 512, 2048 * 512);
  cvt(whh1_bf, whh1, 2048 * 512, 2048 * 512);
  cvt(head_bf, headw, 2502 * 512, 2512 * 512);
  cvt(t1w1_bf, t1w1, 128 * 512, 128 * 512);
  cvt(t1w2_bf, t1w2, 7500 * 128, 7504 * 128);
  cvt(t2w1_bf, t2w1, 32 * 512, 32 * 512);
  cvt(t2w2_bf, t2w2, 40000 * 32, 40000 * 32);

  gather_x_kernel<<<Sv * Bv, 256, 0, stream>>>(inputs, emb, x_bf);
  gather_topics_kernel<<<Bv * Kv, 256, 0, stream>>>(topics, emb, topics_f,
                                                    topics_bf);
  init_kernel<<<64, 256, 0, stream>>>(output0, h0in, c0in, covin, q_buf, h0f,
                                      h1f, c0f, c1f, h0bf, h1bf, cov);
  phi_kernel<<<Bv * Kv, 32, 0, stream>>>(topics_f, Uf, mask, phi);
  // proj_keys [160,512] = topics_bf @ Ua^T : 10x32 tiles = 320 waves
  gemm2_kernel<512, 0><<<40, 256, 0, stream>>>(topics_bf, Ua_bf, topics_bf,
                                               Ua_bf, projk, 10, 32, 512);

  for (int t = 0; t < Sv; ++t) {
    attention_kernel<<<1, 1024, 0, stream>>>(
        t, q_buf, qqA, qq_bf, Wa_bf, projk, va, vab, cov, phi, topics_f, x_bf,
        out + OFF_ATTN + (size_t)t * Bv * Kv);
    // gates0 = x_t @ w_ih0^T + h0 @ w_hh0^T : 2x128 tiles = 256 waves
    gemm2_kernel<1024, 512><<<32, 256, 0, stream>>>(
        x_bf + (size_t)t * Bv * 1024, wih0_bf, h0bf, whh0_bf, gates0, 2, 128,
        2048);
    lstm_elem_kernel<<<64, 256, 0, stream>>>(gates0, bih0, bhh0, c0f, h0f, h0bf,
                                             nullptr, nullptr, nullptr);
    gemm2_kernel<512, 512><<<32, 256, 0, stream>>>(h0bf, wih1_bf, h1bf, whh1_bf,
                                                   gates1, 2, 128, 2048);
    lstm_elem_kernel<<<64, 256, 0, stream>>>(
        gates1, bih1, bhh1, c1f, h1f, h1bf,
        out + OFF_OUTS + (size_t)t * Bv * Hv, outs_bf + (size_t)t * Bv * Hv,
        q_buf);
  }

  // adaptive softmax tails
  gemm2_kernel<512, 0><<<128, 256, 0, stream>>>(outs_bf, t1w1_bf, outs_bf,
                                                t1w1_bf, hid1, 128, 8, 128);
  cvt(hid1_bf, hid1, Nrows * H1v, Nrows * H1v);
  gemm2_kernel<512, 0><<<32, 256, 0, stream>>>(outs_bf, t2w1_bf, outs_bf,
                                               t2w1_bf, hid2, 128, 2, 32);
  cvt(hid2_bf, hid2, Nrows * H2v, Nrows * H2v);

  tgt_kernel<<<256, 256, 0, stream>>>(target, outs_bf, head_bf, hid1_bf,
                                      t1w2_bf, hid2_bf, t2w2_bf, hdt, hdc1,
                                      hdc2, l1t, l2t);
  lse_kernel<512><<<Nrows / 16, 32, 0, stream>>>(outs_bf, head_bf, 2512 / 16,
                                                 C1v + 2, lseh);
  lse_kernel<128><<<Nrows / 16, 32, 0, stream>>>(hid1_bf, t1w2_bf, 7504 / 16,
                                                 CSZ1v, lse1);
  lse_kernel<32><<<Nrows / 16, 32, 0, stream>>>(hid2_bf, t2w2_bf, 40000 / 16,
                                                CSZ2v, lse2);
  final_kernel<<<1, 1024, 0, stream>>>(target, hdt, hdc1, hdc2, l1t, l2t, lseh,
                                       lse1, lse2, out);
  copyfinal_kernel<<<64, 256, 0, stream>>>(h0f, h1f, c0f, c1f, cov, out);
}
